// SoftMoEVelocityBias_8383776162150
// MI455X (gfx1250) — matrix operand order repaired and checked by reading, not compiled, not run on hardware
//
#include <hip/hip_runtime.h>

// CDNA5 (gfx1250) wave32 WMMA types
typedef __bf16 v16bf __attribute__((ext_vector_type(16)));
typedef float  v8f   __attribute__((ext_vector_type(8)));

#define WMMA_BF16(a, b, c) \
    __builtin_amdgcn_wmma_f32_16x16x32_bf16(false, (a), false, (b), (short)0, (c), false, false)

// Problem constants
#define BB 16384
#define DD 512
#define EE 16
#define HH 256
#define ZZ 256

// ---------------------------------------------------------------------------
// Pack h (B x D, f32 row-major) into WMMA A-fragment-packed bf16.
// A 16-bit 16x32 layout: lane<16 -> M=l, K = {0..7, 16..23}; lane>=16 -> M=l-16,
// K = {8..15, 24..31}. Linear j in lane: col = kt*32 + (l<16?0:8) + (j>>3)*16 + (j&7)
// ---------------------------------------------------------------------------
__global__ void pack_a_kernel(const float* __restrict__ src, __bf16* __restrict__ dst,
                              int total) {
    int o = blockIdx.x * 256 + threadIdx.x;
    if (o >= total) return;
    int rt   = o >> 13;          // 16 frags * 512 = 8192 elems per 16-row tile
    int rem  = o & 8191;
    int kt   = rem >> 9;
    int lane = (rem >> 4) & 31;
    int j    = rem & 15;
    int row  = rt * 16 + (lane & 15);
    int col  = kt * 32 + ((lane < 16) ? 0 : 8) + ((j >> 3) << 4) + (j & 7);
    dst[o] = (__bf16)src[row * DD + col];
}

// ---------------------------------------------------------------------------
// Pack a (batched) K x N f32 row-major matrix into WMMA B-fragment-packed bf16.
// B 16-bit 32x16 layout: lane<16 -> N=l, K = {0..7, 16..23}; lane>=16 -> N=l-16, K = {8..15, 24..31}
// ---------------------------------------------------------------------------
__global__ void pack_b_kernel(const float* __restrict__ src, __bf16* __restrict__ dst,
                              int K, int N, int total) {
    int o = blockIdx.x * 256 + threadIdx.x;
    if (o >= total) return;
    int per  = K * N;
    int b    = o / per;
    int rem  = o - b * per;
    int nnt  = N >> 4;
    int f    = rem >> 9;
    int lane = (rem >> 4) & 31;
    int j    = rem & 15;
    int kt   = f / nnt;
    int nt   = f - kt * nnt;
    int k    = kt * 32 + ((lane < 16) ? 0 : 8) + ((j >> 3) << 4) + (j & 7);
    int n    = nt * 16 + (lane & 15);
    dst[o] = (__bf16)src[b * per + k * N + n];
}

// ---------------------------------------------------------------------------
// Pack b2 (E=16 x Z=256, f32) as a zero-padded K=32 WMMA B operand:
// K index = expert e (real for K<16, zero for K>=16). 16 n-tiles of 512 bf16.
// ---------------------------------------------------------------------------
__global__ void pack_b2_kernel(const float* __restrict__ b2, __bf16* __restrict__ dst,
                               int total) {
    int o = blockIdx.x * 256 + threadIdx.x;
    if (o >= total) return;
    int nt   = o >> 9;
    int lane = (o >> 4) & 31;
    int j    = o & 15;
    int n    = nt * 16 + (lane & 15);
    float v  = (j < 8) ? b2[(((lane < 16) ? 0 : 8) + j) * ZZ + n] : 0.0f;   // K = 8h + j (j<8) real, else pad
    dst[o] = (__bf16)v;
}

// ---------------------------------------------------------------------------
// Async copy: global -> LDS, 16 bytes per lane, tracked by ASYNCcnt.
// ---------------------------------------------------------------------------
__device__ __forceinline__ void async_copy_b128(unsigned lds_off, const void* gptr) {
    asm volatile("global_load_async_to_lds_b128 %0, %1, off"
                 :: "v"(lds_off), "v"(gptr) : "memory");
}
__device__ __forceinline__ void wait_asynccnt0() {
    asm volatile("s_wait_asynccnt 0x0" ::: "memory");
}

// ---------------------------------------------------------------------------
// Main fused kernel. 256 threads = 8 waves. Block covers 64 rows (4 row-tiles).
// Wave w owns columns [w*32, w*32+32) of both H (hidden) and Z (output).
// ---------------------------------------------------------------------------
__global__ __launch_bounds__(256) void moe_kernel(
    const __bf16* __restrict__ hA,    // A-packed h
    const __bf16* __restrict__ phiP,  // B-packed phi (K=512, N=16)
    const __bf16* __restrict__ w1P,   // B-packed W1  (E x [K=512, N=256])
    const __bf16* __restrict__ w2P,   // B-packed W2  (E x [K=256, N=256])
    const __bf16* __restrict__ b2P,   // B-packed zero-padded b2 (K=32, N=256)
    const float* __restrict__ b1,     // (E, H) f32
    float* __restrict__ out)          // (B, Z) f32
{
    __shared__ __align__(32) __bf16 shA[4 * 16 * 512];    // 64 KB: h A-frags
    __shared__ __align__(32) __bf16 shHid[4 * 8 * 512];   // 32 KB: hid' A-frags
    // 4 KB region: first logits f32[64][16], then reused as g A-frags bf16[4][512]
    __shared__ __align__(32) unsigned char shLogGA[4096];
    __shared__ __align__(16) float shGT[16 * 64];         // 4 KB: g transposed [e][row]

    float* shLog = (float*)shLogGA;
    __bf16* shGA = (__bf16*)shLogGA;

    const int tid  = threadIdx.x;
    const int lane = tid & 31;
    const int wave = tid >> 5;
    const int blk  = blockIdx.x;
    const int nlo  = lane & 15;
    const int mbase = (lane < 16) ? 0 : 8;

    // ---- stage 0: async copy of the block's packed h tiles (64 KB) ----
    {
        const uint4* src = (const uint4*)(hA + (size_t)blk * (4 * 16 * 512));
        const unsigned lds_base = (unsigned)(size_t)(void*)shA;
        #pragma unroll
        for (int i = 0; i < 16; ++i)
            async_copy_b128(lds_base + (tid + 256 * i) * 16, src + tid + 256 * i);
        wait_asynccnt0();
    }
    __syncthreads();

    // ---- stage 1: logits = h @ phi  (waves 0..3, one row-tile each) ----
    if (wave < 4) {
        const int rt = wave;
        v8f acc = (v8f)0.0f;
        #pragma unroll
        for (int k = 0; k < 16; ++k) {
            v16bf a = *(const v16bf*)(shA + (rt * 16 + k) * 512 + lane * 16);
            v16bf b = *(const v16bf*)(phiP + k * 512 + lane * 16);
            acc = WMMA_BF16(a, b, acc);
        }
        #pragma unroll
        for (int r = 0; r < 8; ++r)
            shLog[(rt * 16 + mbase + r) * 16 + nlo] = acc[r];
    }
    __syncthreads();

    // ---- stage 1b: per-row softmax over 16 experts (row = tid for tid<64).
    // Emits g transposed (shGT[e][row]) and g as zero-padded K=32 A-fragments
    // (shGA, aliasing shLog: per-wave DS ordering makes read-then-write safe).
    if (tid < 64) {
        const int rt = tid >> 4;
        const int m  = tid & 15;
        float v[16];
        float mx = -3.4e38f;
        #pragma unroll
        for (int e = 0; e < 16; ++e) { v[e] = shLog[tid * 16 + e]; mx = fmaxf(mx, v[e]); }
        float s = 0.0f;
        #pragma unroll
        for (int e = 0; e < 16; ++e) { v[e] = __expf(v[e] - mx); s += v[e]; }
        float inv = 1.0f / s;
        #pragma unroll
        for (int e = 0; e < 16; ++e) {
            float g = v[e] * inv;
            shGT[e * 64 + tid] = g;
            // A-frag: K=e -> lane' = m + (e>=8 ? 16:0), j = e&7
            shGA[rt * 512 + (m + ((e >= 8) ? 16 : 0)) * 16 + (e & 7)] = (__bf16)g;
        }
        #pragma unroll
        for (int j = 8; j < 16; ++j) {   // zero-pad K = 16..31
            shGA[rt * 512 + m * 16 + j] = (__bf16)0.0f;
            shGA[rt * 512 + (m + 16) * 16 + j] = (__bf16)0.0f;
        }
    }
    __syncthreads();

    // ---- persistent output accumulators: 2 ntiles x 4 rowtiles ----
    v8f outacc[2][4];
    #pragma unroll
    for (int nt = 0; nt < 2; ++nt)
        #pragma unroll
        for (int rt = 0; rt < 4; ++rt)
            outacc[nt][rt] = (v8f)0.0f;

    const int colbase = wave * 32;

    for (int e = 0; e < EE; ++e) {
        // ---- stage 2: hid = h @ W1[e] over the wave's 32 H-columns ----
        v8f hacc[2][4];
        #pragma unroll
        for (int nt = 0; nt < 2; ++nt)
            #pragma unroll
            for (int rt = 0; rt < 4; ++rt)
                hacc[nt][rt] = (v8f)0.0f;

        const __bf16* w1e = w1P + (size_t)e * (16 * 16 * 512);
        for (int k = 0; k < 16; ++k) {
            v16bf a[4];
            #pragma unroll
            for (int rt = 0; rt < 4; ++rt)
                a[rt] = *(const v16bf*)(shA + (rt * 16 + k) * 512 + lane * 16);
            #pragma unroll
            for (int nt = 0; nt < 2; ++nt) {
                const int gnt = wave * 2 + nt;
                v16bf b = *(const v16bf*)(w1e + (k * 16 + gnt) * 512 + lane * 16);
                #pragma unroll
                for (int rt = 0; rt < 4; ++rt)
                    hacc[nt][rt] = WMMA_BF16(a[rt], b, hacc[nt][rt]);
            }
        }

        // ---- fused bias + relu + combine-weight scale + bf16, scatter into
        //      LDS in A-fragment-packed layout (K dim = H) ----
        #pragma unroll
        for (int nt = 0; nt < 2; ++nt) {
            const int hcol   = colbase + nt * 16 + nlo;
            const float bias = b1[e * HH + hcol];
            const int kk     = hcol >> 5;
            const int klocal = hcol & 31;
            const int hi     = (klocal >> 3) & 1;
            const int jj     = (klocal & 7) + ((klocal >> 4) << 3);
            #pragma unroll
            for (int rt = 0; rt < 4; ++rt) {
                const float* gp = shGT + e * 64 + rt * 16 + mbase;
                float4 gv0 = *(const float4*)(gp);
                float4 gv1 = *(const float4*)(gp + 4);
                #pragma unroll
                for (int r = 0; r < 8; ++r) {
                    const int m = mbase + r;
                    const float g = (r < 4) ? (&gv0.x)[r] : (&gv1.x)[r - 4];
                    float val = fmaxf(hacc[nt][rt][r] + bias, 0.0f) * g;
                    shHid[(rt * 8 + kk) * 512 + (m + hi * 16) * 16 + jj] = (__bf16)val;
                }
            }
        }
        __syncthreads();

        // ---- stage 3: out += hid' @ W2[e]  (K = 256) ----
        const __bf16* w2e = w2P + (size_t)e * (8 * 16 * 512);
        for (int kk = 0; kk < 8; ++kk) {
            v16bf a[4];
            #pragma unroll
            for (int rt = 0; rt < 4; ++rt)
                a[rt] = *(const v16bf*)(shHid + (rt * 8 + kk) * 512 + lane * 16);
            #pragma unroll
            for (int nt = 0; nt < 2; ++nt) {
                const int gnt = wave * 2 + nt;
                v16bf b = *(const v16bf*)(w2e + (kk * 16 + gnt) * 512 + lane * 16);
                #pragma unroll
                for (int rt = 0; rt < 4; ++rt)
                    outacc[nt][rt] = WMMA_BF16(a[rt], b, outacc[nt][rt]);
            }
        }
        __syncthreads();   // protect shHid before next expert overwrites it
    }

    // ---- epilogue: out += g @ b2 via WMMA (K=32 zero-padded), then store ----
    #pragma unroll
    for (int nt = 0; nt < 2; ++nt) {
        v16bf b = *(const v16bf*)(b2P + (wave * 2 + nt) * 512 + lane * 16);
        #pragma unroll
        for (int rt = 0; rt < 4; ++rt) {
            v16bf ga = *(const v16bf*)(shGA + rt * 512 + lane * 16);
            outacc[nt][rt] = WMMA_BF16(ga, b, outacc[nt][rt]);
        }
    }

    #pragma unroll
    for (int nt = 0; nt < 2; ++nt) {
        const int z = colbase + nt * 16 + nlo;
        #pragma unroll
        for (int rt = 0; rt < 4; ++rt) {
            #pragma unroll
            for (int r = 0; r < 8; ++r) {
                const int row = blk * 64 + rt * 16 + mbase + r;
                out[row * ZZ + z] = outacc[nt][rt][r];
            }
        }
    }
}

// ---------------------------------------------------------------------------
extern "C" void kernel_launch(void* const* d_in, const int* in_sizes, int n_in,
                              void* d_out, int out_size, void* d_ws, size_t ws_size,
                              hipStream_t stream) {
    const float* h   = (const float*)d_in[0];
    const float* phi = (const float*)d_in[1];
    const float* W1  = (const float*)d_in[2];
    const float* b1  = (const float*)d_in[3];
    const float* W2  = (const float*)d_in[4];
    const float* b2  = (const float*)d_in[5];
    float* out = (float*)d_out;

    // Workspace layout (bf16 packed operands), ~22 MB total
    __bf16* hA   = (__bf16*)d_ws;                 // 16384*512
    __bf16* phiP = hA + (size_t)BB * DD;          // 512*16
    __bf16* w1P  = phiP + (size_t)DD * EE;        // 16*512*256
    __bf16* w2P  = w1P + (size_t)EE * DD * HH;    // 16*256*256
    __bf16* b2P  = w2P + (size_t)EE * HH * ZZ;    // 16 n-tiles * 512 (K=32 padded)

    const int totH  = BB * DD;            // 8388608
    const int totPh = DD * EE;            // 8192
    const int totW1 = EE * DD * HH;       // 2097152
    const int totW2 = EE * HH * ZZ;       // 1048576
    const int totB2 = 16 * 512;           // 8192

    pack_a_kernel<<<totH / 256, 256, 0, stream>>>(h, hA, totH);
    pack_b_kernel<<<totPh / 256, 256, 0, stream>>>(phi, phiP, DD, EE, totPh);
    pack_b_kernel<<<totW1 / 256, 256, 0, stream>>>(W1, w1P, DD, HH, totW1);
    pack_b_kernel<<<totW2 / 256, 256, 0, stream>>>(W2, w2P, HH, ZZ, totW2);
    pack_b2_kernel<<<totB2 / 256, 256, 0, stream>>>(b2, b2P, totB2);

    moe_kernel<<<BB / 64, 256, 0, stream>>>(hA, phiP, w1P, w2P, b2P, b1, out);
}
